// MultiGraphGAT_11510512354047
// MI455X (gfx1250) — matrix operand order repaired and checked by reading, not compiled, not run on hardware
//
#include <hip/hip_runtime.h>

// ---------------------------------------------------------------------------
// MultiGraphGAT for MI455X (gfx1250): 2-layer GAT over 2 graphs.
//   GEMMs       -> LDS-staged v_wmma_f32_16x16x32_f16, 8 waves/block,
//                  128x64 block tile, 4 WMMA tiles per wave per K-step.
//                  Staging: clause-batched global float4 loads -> cvt -> LDS.
//   softmax     -> 3-pass segment softmax with float atomic max/add
//   aggregation -> wave-per-edge float4 gather + global_atomic_add_f32
// ---------------------------------------------------------------------------

typedef __attribute__((ext_vector_type(16))) _Float16 v16h;
typedef __attribute__((ext_vector_type(8)))  float    v8f;

#define N_NODES 50000
#define N_EDGES 800000
#define N_GRAPHS 2
#define NEG_SLOPE 0.2f

#define TM 128   // block tile rows (8 waves x 16)
#define TN 64    // block tile cols (4 WMMA tiles per wave)
#define PA 40    // LDS pitch (halfs) for A panel (32 + pad)
#define PB 40    // LDS pitch (halfs) for B^T panel

// ---------------------------------------------------------------------------
// GEMM: C[M,N] = A[M,K] * B[K,N], f32 in / f32 out, f16 WMMA inside.
// Fragment layouts (wave32); C/D per CDNA5 ISA 7.12.2, A/B use contiguous k:
//  A 16x32 f16:  lane m=(L&15), hi=(L>>4); vector elem e = K 16*hi+e,
//                same k order as B;  (two contiguous 16B LDS reads)
//  B 32x16 f16:  lane n=(L&15), hi=(L>>4); elem e = K 16*hi+e
//                (one contiguous 32B read from transposed LDS panel)
//  C/D 16x16 f32: VGPR r -> M = r + 8*hi, N = (L&15).
// OOB rows are handled by CLAMPING the A row index (A row r only feeds C
// row r, and C stores for rows >= M are guarded), so staging is branch-free
// and all global loads issue as one clause.
// ---------------------------------------------------------------------------
__global__ __launch_bounds__(256)
void gat_gemm_wmma(const float* __restrict__ A, const float* __restrict__ B,
                   float* __restrict__ C, int M, int N, int K) {
    __shared__ _Float16 ldsA[TM * PA];
    __shared__ _Float16 ldsBT[TN * PB];

    const int tid  = threadIdx.x;
    const int wave = tid >> 5;
    const int lane = tid & 31;
    const int lm   = lane & 15;
    const int hi   = lane >> 4;
    const int rowBase = blockIdx.x * TM;
    const int colBase = blockIdx.y * TN;

    // Per-thread staging coordinates (fixed across the K loop).
    const int arow = tid >> 3;             // 0..31 (stepped by +32, 4x)
    const int acol = (tid & 7) * 4;        // 0,4,..,28
    const int brow = tid >> 4;             // 0..15 (stepped by +16, 2x)
    const int bcol = (tid & 15) * 4;       // 0,4,..,60

    v8f acc[4] = {{0.f}, {0.f}, {0.f}, {0.f}};

    for (int kb = 0; kb < K; kb += 32) {
        // ---- gather staging data (branch-free, single load clause) ----
        float4 va[TM / 32], vb[2];
#pragma unroll
        for (int it = 0; it < TM / 32; ++it) {
            int gr = rowBase + it * 32 + arow;
            gr = gr < M ? gr : M - 1;      // clamp, never branches
            va[it] = *(const float4*)&A[(size_t)gr * K + kb + acol];
        }
#pragma unroll
        for (int it = 0; it < 2; ++it) {
            const int k = it * 16 + brow;
            vb[it] = *(const float4*)&B[(size_t)(kb + k) * N + colBase + bcol];
        }

        __syncthreads();   // WAR: previous iteration's fragment reads done

        // ---- convert + store panels to LDS ----
#pragma unroll
        for (int it = 0; it < TM / 32; ++it) {
            _Float16* dp = &ldsA[(it * 32 + arow) * PA + acol];
            dp[0] = (_Float16)va[it].x; dp[1] = (_Float16)va[it].y;
            dp[2] = (_Float16)va[it].z; dp[3] = (_Float16)va[it].w;
        }
#pragma unroll
        for (int it = 0; it < 2; ++it) {
            const int k = it * 16 + brow;
            ldsBT[(bcol + 0) * PB + k] = (_Float16)vb[it].x;
            ldsBT[(bcol + 1) * PB + k] = (_Float16)vb[it].y;
            ldsBT[(bcol + 2) * PB + k] = (_Float16)vb[it].z;
            ldsBT[(bcol + 3) * PB + k] = (_Float16)vb[it].w;
        }
        __syncthreads();

        // ---- A fragment for this wave (reused across 4 column tiles) ----
        const _Float16* __restrict__ ap = &ldsA[(wave * 16 + lm) * PA + 16 * hi];
        v16h a;
#pragma unroll
        for (int i = 0; i < 8; ++i) { a[i] = ap[i]; a[8 + i] = ap[8 + i]; }

#pragma unroll
        for (int t = 0; t < 4; ++t) {
            const _Float16* __restrict__ bp = &ldsBT[(t * 16 + lm) * PB + 16 * hi];
            v16h b;
#pragma unroll
            for (int i = 0; i < 16; ++i) b[i] = bp[i];
            acc[t] = __builtin_amdgcn_wmma_f32_16x16x32_f16(
                /*neg_a=*/false, a, /*neg_b=*/false, b,
                /*c_mod=*/(short)0, acc[t], /*reuse_a=*/false, /*reuse_b=*/false);
        }
    }

    // ---- store C (coalesced across lanes for each r) ----
    const int m0 = rowBase + wave * 16 + 8 * hi;
#pragma unroll
    for (int t = 0; t < 4; ++t) {
        const int cn = colBase + t * 16 + lm;
#pragma unroll
        for (int r = 0; r < 8; ++r) {
            const int m = m0 + r;
            if (m < M) C[(size_t)m * N + cn] = acc[t][r];
        }
    }
}

// ---------------------------------------------------------------------------
// al_src[n,h] = sum_c h[n,h*C+c]*a_src[h,c]  (same for a_dst), float4 inner
// ---------------------------------------------------------------------------
template <int H, int C>
__global__ __launch_bounds__(256)
void gat_attn_logits(const float* __restrict__ h,
                     const float* __restrict__ a_src, const float* __restrict__ a_dst,
                     float* __restrict__ als, float* __restrict__ ald, int Nn) {
    const int idx = blockIdx.x * blockDim.x + threadIdx.x;   // over Nn*H
    if (idx >= Nn * H) return;
    const int n = idx / H, hh = idx - n * H;
    const float4* __restrict__ hp = (const float4*)(h + (size_t)n * H * C + hh * C);
    const float4* __restrict__ as = (const float4*)(a_src + hh * C);
    const float4* __restrict__ ad = (const float4*)(a_dst + hh * C);
    float ss = 0.f, sd = 0.f;
#pragma unroll 4
    for (int c = 0; c < C / 4; ++c) {
        const float4 v = hp[c], s4 = as[c], d4 = ad[c];
        ss += v.x * s4.x + v.y * s4.y + v.z * s4.z + v.w * s4.w;
        sd += v.x * d4.x + v.y * d4.y + v.z * d4.z + v.w * d4.w;
    }
    als[idx] = ss;
    ald[idx] = sd;
}

__global__ __launch_bounds__(256)
void gat_fill(float* __restrict__ p, float v, long long n) {
    const long long i = (long long)blockIdx.x * blockDim.x + threadIdx.x;
    if (i < n) p[i] = v;
}

// Sign-split float atomic max (memory must be initialized to -inf).
__device__ inline void atomicMaxF(float* addr, float val) {
    if (val >= 0.f) atomicMax((int*)addr, __float_as_int(val));
    else            atomicMin((unsigned int*)addr, __float_as_uint(val));
}

__device__ inline float leaky(float v) { return v > 0.f ? v : v * NEG_SLOPE; }

// ---------------------------------------------------------------------------
// Pass 1: per-dst segment max of leaky_relu(al_src[src]+al_dst[dst])
// ---------------------------------------------------------------------------
template <int H>
__global__ __launch_bounds__(256)
void gat_edge_max(const int* __restrict__ ei, int E, int Nn,
                  const float* __restrict__ als, const float* __restrict__ ald,
                  float* __restrict__ m) {
    const int e = blockIdx.x * blockDim.x + threadIdx.x;
    if (e >= E + Nn) return;
    const int src = (e < E) ? ei[e]     : (e - E);
    const int dst = (e < E) ? ei[E + e] : (e - E);
#pragma unroll
    for (int hh = 0; hh < H; ++hh) {
        const float v = leaky(als[src * H + hh] + ald[dst * H + hh]);
        atomicMaxF(&m[dst * H + hh], v);
    }
}

// ---------------------------------------------------------------------------
// Pass 2: per-dst sum of exp(e - m[dst])
// ---------------------------------------------------------------------------
template <int H>
__global__ __launch_bounds__(256)
void gat_edge_sum(const int* __restrict__ ei, int E, int Nn,
                  const float* __restrict__ als, const float* __restrict__ ald,
                  const float* __restrict__ m, float* __restrict__ s) {
    const int e = blockIdx.x * blockDim.x + threadIdx.x;
    if (e >= E + Nn) return;
    const int src = (e < E) ? ei[e]     : (e - E);
    const int dst = (e < E) ? ei[E + e] : (e - E);
#pragma unroll
    for (int hh = 0; hh < H; ++hh) {
        const float v = leaky(als[src * H + hh] + ald[dst * H + hh]);
        atomicAdd(&s[dst * H + hh], __expf(v - m[dst * H + hh]));
    }
}

// ---------------------------------------------------------------------------
// Pass 3: acc[dst,:] += alpha * h[src,:].  One wave32 per edge; each lane
// owns float4 chunks so gathers are 512B-contiguous per wave.
// ---------------------------------------------------------------------------
template <int H, int C>
__global__ __launch_bounds__(256)
void gat_edge_aggregate(const int* __restrict__ ei, int E, int Nn,
                        const float* __restrict__ h,
                        const float* __restrict__ als, const float* __restrict__ ald,
                        const float* __restrict__ m, const float* __restrict__ s,
                        float* __restrict__ acc) {
    const int wave = (blockIdx.x * blockDim.x + threadIdx.x) >> 5;
    const int lane = threadIdx.x & 31;
    if (wave >= E + Nn) return;
    const int src = (wave < E) ? ei[wave]     : (wave - E);
    const int dst = (wave < E) ? ei[E + wave] : (wave - E);

    float alpha[H];
#pragma unroll
    for (int hh = 0; hh < H; ++hh) {
        const float v = leaky(als[src * H + hh] + ald[dst * H + hh]);
        alpha[hh] = __expf(v - m[dst * H + hh]) / s[dst * H + hh];
    }

    constexpr int F  = H * C;
    constexpr int CH = F / 4;          // float4 chunks per edge
    const float4* __restrict__ hs4 = (const float4*)(h + (size_t)src * F);
    float* __restrict__        ad  = acc + (size_t)dst * F;
#pragma unroll
    for (int j = lane; j < CH; j += 32) {
        const float4 v  = hs4[j];
        const float  al = alpha[(4 * j) / C];
        atomicAdd(&ad[4 * j + 0], al * v.x);
        atomicAdd(&ad[4 * j + 1], al * v.y);
        atomicAdd(&ad[4 * j + 2], al * v.z);
        atomicAdd(&ad[4 * j + 3], al * v.w);
    }
}

// out[n,f] = elu(acc[n,f] + bias[f]), float4
template <int F>
__global__ __launch_bounds__(256)
void gat_bias_elu(const float* __restrict__ acc, const float* __restrict__ bias,
                  float* __restrict__ out, int Nn) {
    const long long i = (long long)blockIdx.x * blockDim.x + threadIdx.x;
    if (i >= (long long)Nn * (F / 4)) return;
    const float4 a4 = ((const float4*)acc)[i];
    const float4 b4 = ((const float4*)bias)[(int)(i % (F / 4))];
    float4 o;
    float v;
    v = a4.x + b4.x; o.x = v > 0.f ? v : (__expf(v) - 1.f);
    v = a4.y + b4.y; o.y = v > 0.f ? v : (__expf(v) - 1.f);
    v = a4.z + b4.z; o.z = v > 0.f ? v : (__expf(v) - 1.f);
    v = a4.w + b4.w; o.w = v > 0.f ? v : (__expf(v) - 1.f);
    ((float4*)out)[i] = o;
}

// ---------------------------------------------------------------------------
// Host-side orchestration
// ---------------------------------------------------------------------------
static inline int cdiv(long long a, long long b) { return (int)((a + b - 1) / b); }

extern "C" void kernel_launch(void* const* d_in, const int* in_sizes, int n_in,
                              void* d_out, int out_size, void* d_ws, size_t ws_size,
                              hipStream_t stream) {
    (void)in_sizes; (void)n_in; (void)out_size; (void)ws_size;
    const float* x_all  = (const float*)d_in[0];  // [G, N, 128]
    const int*   ei_all = (const int*)  d_in[1];  // [G, 2, E]
    const float* W1     = (const float*)d_in[2];  // [G, 128, 256]
    const float* a1s    = (const float*)d_in[3];  // [G, 4, 64]
    const float* a1d    = (const float*)d_in[4];
    const float* b1     = (const float*)d_in[5];  // [G, 256]
    const float* W2     = (const float*)d_in[6];  // [G, 256, 128]
    const float* a2s    = (const float*)d_in[7];  // [G, 1, 128]
    const float* a2d    = (const float*)d_in[8];
    const float* b2     = (const float*)d_in[9];  // [G, 128]
    float* out = (float*)d_out;                   // [G*N, 128]

    const int Nn = N_NODES, E = N_EDGES;

    // Workspace layout (floats):
    //   bufH  [N*256]  : h1; reused as h2 [N*128] + acc2 [N*128]
    //   bufA1 [N*256]  : layer-1 accumulator -> ELU'd x2
    //   als/ald/m/s [N*4] each
    float* ws    = (float*)d_ws;
    float* bufH  = ws;
    float* bufA1 = bufH  + (size_t)Nn * 256;
    float* als   = bufA1 + (size_t)Nn * 256;
    float* ald   = als   + (size_t)Nn * 4;
    float* mbuf  = ald   + (size_t)Nn * 4;
    float* sbuf  = mbuf  + (size_t)Nn * 4;
    float* h2    = bufH;                      // reuse
    float* acc2  = bufH + (size_t)Nn * 128;   // reuse

    const int edge_threads = E + Nn;                       // 850000
    const int edge_blocks  = cdiv(edge_threads, 256);
    const int agg_blocks   = cdiv((long long)edge_threads * 32, 256);

    for (int g = 0; g < N_GRAPHS; ++g) {
        const float* x  = x_all  + (size_t)g * Nn * 128;
        const int*   ei = ei_all + (size_t)g * 2 * E;

        // ---- Layer 1: h1 = x @ W1  (50000x128 @ 128x256) ----
        {
            dim3 grid(cdiv(Nn, TM), 256 / TN);
            gat_gemm_wmma<<<grid, 256, 0, stream>>>(x, W1 + (size_t)g * 128 * 256,
                                                    bufH, Nn, 256, 128);
        }
        gat_attn_logits<4, 64><<<cdiv((long long)Nn * 4, 256), 256, 0, stream>>>(
            bufH, a1s + (size_t)g * 4 * 64, a1d + (size_t)g * 4 * 64, als, ald, Nn);

        gat_fill<<<cdiv((long long)Nn * 4, 256), 256, 0, stream>>>(mbuf, -INFINITY, (long long)Nn * 4);
        gat_fill<<<cdiv((long long)Nn * 4, 256), 256, 0, stream>>>(sbuf, 0.f, (long long)Nn * 4);
        gat_fill<<<cdiv((long long)Nn * 256, 256), 256, 0, stream>>>(bufA1, 0.f, (long long)Nn * 256);

        gat_edge_max<4><<<edge_blocks, 256, 0, stream>>>(ei, E, Nn, als, ald, mbuf);
        gat_edge_sum<4><<<edge_blocks, 256, 0, stream>>>(ei, E, Nn, als, ald, mbuf, sbuf);
        gat_edge_aggregate<4, 64><<<agg_blocks, 256, 0, stream>>>(ei, E, Nn, bufH,
                                                                  als, ald, mbuf, sbuf, bufA1);
        gat_bias_elu<256><<<cdiv((long long)Nn * 64, 256), 256, 0, stream>>>(
            bufA1, b1 + (size_t)g * 256, bufA1, Nn);   // in-place ELU -> x2

        // ---- Layer 2: h2 = x2 @ W2  (50000x256 @ 256x128) ----
        {
            dim3 grid(cdiv(Nn, TM), 128 / TN);
            gat_gemm_wmma<<<grid, 256, 0, stream>>>(bufA1, W2 + (size_t)g * 256 * 128,
                                                    h2, Nn, 128, 256);
        }
        gat_attn_logits<1, 128><<<cdiv((long long)Nn, 256), 256, 0, stream>>>(
            h2, a2s + (size_t)g * 128, a2d + (size_t)g * 128, als, ald, Nn);

        gat_fill<<<cdiv((long long)Nn, 256), 256, 0, stream>>>(mbuf, -INFINITY, Nn);
        gat_fill<<<cdiv((long long)Nn, 256), 256, 0, stream>>>(sbuf, 0.f, Nn);
        gat_fill<<<cdiv((long long)Nn * 128, 256), 256, 0, stream>>>(acc2, 0.f, (long long)Nn * 128);

        gat_edge_max<1><<<edge_blocks, 256, 0, stream>>>(ei, E, Nn, als, ald, mbuf);
        gat_edge_sum<1><<<edge_blocks, 256, 0, stream>>>(ei, E, Nn, als, ald, mbuf, sbuf);
        gat_edge_aggregate<1, 128><<<agg_blocks, 256, 0, stream>>>(ei, E, Nn, h2,
                                                                   als, ald, mbuf, sbuf, acc2);
        gat_bias_elu<128><<<cdiv((long long)Nn * 32, 256), 256, 0, stream>>>(
            acc2, b2 + (size_t)g * 128, out + (size_t)g * Nn * 128, Nn);
    }
}
